// LinearCDE_56495999812316
// MI455X (gfx1250) — matrix operand order repaired and checked by reading, not compiled, not run on hardware
//
#include <hip/hip_runtime.h>
#include <hip/hip_bf16.h>

// ---------------------------------------------------------------------------
// LinearCDE on MI455X (gfx1250, wave32, WMMA).
//
// Phases per launch:
//   1) y0 init (tiny GEMV) -> ystate + out[:,0,:]
//   2) Bs GEMV (small)
//   3) bf16 conversion of inp_t (padded K=96) and vf_A_W (padded K=96)
//   4) 8 chunks of 64 timesteps:
//        a) WMMA bf16 GEMM: As_chunk[(b,s), 16384] = inp_t @ W_A^T  (33.5 MB,
//           stays resident in the 192MB L2). 2 m-tiles per block so each
//           B fragment load feeds 2 wmmas.
//        b) persistent scan: 8 workgroups (1/batch), 512 threads, 4-way
//           j-split reduced with __shfl_xor (in-wave), ping-pong y buffer,
//           ONE barrier per step, prefetch of step t+1.
// ---------------------------------------------------------------------------

typedef __attribute__((ext_vector_type(16))) __bf16 v16bf;
typedef __attribute__((ext_vector_type(8)))  float  v8f;

#define DT_V   0.025f
#define CLIP_V 1.0e11f

constexpr int Bx  = 8;
constexpr int S   = 512;
constexpr int D   = 64;
constexpr int H   = 128;
constexpr int SM1 = 511;    // S-1 scan steps
constexpr int KP  = 96;     // K=65 padded to 3 bf16 WMMA k-tiles of 32
constexpr int N   = 16384;  // H*H
constexpr int TS  = 64;     // timesteps per chunk
constexpr int NCH = 8;      // ceil(SM1/TS)

// round-to-nearest-even float -> bf16 (raw u16)
__device__ inline unsigned short f2bf(float f) {
  union { float f; unsigned u; } x; x.f = f;
  unsigned u = x.u;
  unsigned r = u + 0x7FFFu + ((u >> 16) & 1u);
  return (unsigned short)(r >> 16);
}

// y0 = X[:,0] @ init_W^T + init_b ; also writes out[:,0,:]
__global__ void k_init_y0(const float* __restrict__ X,
                          const float* __restrict__ W,
                          const float* __restrict__ bvec,
                          float* __restrict__ ystate,
                          float* __restrict__ out) {
  int b = blockIdx.x, h = threadIdx.x;
  const float* x0 = X + (size_t)b * S * D;
  const float* w  = W + h * D;
  float acc = bvec[h];
#pragma unroll
  for (int d = 0; d < D; d += 4) {
    float4 xv = *(const float4*)(x0 + d);
    float4 wv = *(const float4*)(w + d);
    acc += xv.x * wv.x + xv.y * wv.y + xv.z * wv.z + xv.w * wv.w;
  }
  ystate[b * H + h] = acc;
  out[(size_t)b * S * H + h] = acc;
}

// Bs[b,s,h] = DT * (WB[h,0] + sum_d X[b,s+1,d] * WB[h,d+1])
__global__ void k_bs(const float* __restrict__ X,
                     const float* __restrict__ WB,
                     float* __restrict__ Bs) {
  int idx = blockIdx.x * blockDim.x + threadIdx.x;
  if (idx >= Bx * SM1 * H) return;
  int h  = idx % H;
  int bs = idx / H;
  int s  = bs % SM1;
  int b  = bs / SM1;
  const float* w  = WB + h * (D + 1);
  const float* xr = X + ((size_t)b * S + (s + 1)) * D;
  float acc = w[0];
  for (int d = 0; d < D; d++) acc += xr[d] * w[d + 1];
  Bs[idx] = DT_V * acc;
}

// inp_t in bf16, padded to KP=96, row s==511 zeroed (pad row for last chunk)
__global__ void k_conv_inp(const float* __restrict__ X,
                           unsigned short* __restrict__ inpb) {
  int idx = blockIdx.x * blockDim.x + threadIdx.x;   // over Bx*S*KP
  int k   = idx % KP;
  int row = idx / KP;
  int s   = row % S;
  int b   = row / S;
  float v = 0.f;
  if (s < SM1) {
    if (k == 0)      v = DT_V;
    else if (k <= D) v = DT_V * X[((size_t)b * S + (s + 1)) * D + (k - 1)];
  }
  inpb[idx] = f2bf(v);
}

// vf_A_W in bf16, padded to KP=96
__global__ void k_conv_W(const float* __restrict__ WA,
                         unsigned short* __restrict__ WAb) {
  int idx = blockIdx.x * blockDim.x + threadIdx.x;   // over N*KP
  int k = idx % KP;
  int o = idx / KP;
  float v = (k < D + 1) ? WA[(size_t)o * (D + 1) + k] : 0.f;
  WAb[idx] = f2bf(v);
}

// bit-cast two uint4 loads into a v16bf WMMA fragment
union F16x16 { struct { uint4 lo, hi; } u; v16bf v; };

// As_chunk[(b*64+s_local) * 16384 + o] = sum_k inp_t[row,k] * W_A[o,k]
// GEMM M=512 (rows = (b, s_local)), N=16384, K=96 (3 bf16 k-tiles).
// 2 m-tiles per block: each B fragment load feeds two wmmas.
// Per-lane fragment addressing follows ISA 7.12.2 VGPR striping:
//   A 16x32 bf16: lane (16g+m): halves = K[kt*32 + g*8 .. +7], K[+16+g*8 ..]
//   B 32x16 bf16: lane (16g+n): halves = K[kt*32 + g*8 .. +7], K[+16+g*8 ..]
__global__ void __launch_bounds__(128)
k_gemm_As(const unsigned short* __restrict__ inpb,
          const unsigned short* __restrict__ WAb,
          float* __restrict__ As, int chunk) {
  int mpair = blockIdx.x & 15;   // 16 pairs of m-tiles (32 rows each)
  int nblk  = blockIdx.x >> 4;   // 64 n-blocks of 256 cols
  int wave  = threadIdx.x >> 5;
  int lane  = threadIdx.x & 31;
  int m0 = mpair * 32;
  int g  = lane >> 4;            // half-wave group
  int nn = lane & 15;            // M row (A) / N col (B,C) within tile

  int b     = m0 / TS;           // 32-row pair never crosses a batch (64%32==0)
  int srow0 = b * S + chunk * TS + (m0 % TS);

  // A fragments (reused across all N): 2 m-tiles x 3 k-tiles
  F16x16 afr[2][3];
#pragma unroll
  for (int mt = 0; mt < 2; mt++) {
    const unsigned short* arow = inpb + (size_t)(srow0 + mt * 16 + nn) * KP;
#pragma unroll
    for (int kt = 0; kt < 3; kt++) {
      afr[mt][kt].u.lo = *(const uint4*)(arow + kt * 32 + g * 8);
      afr[mt][kt].u.hi = *(const uint4*)(arow + kt * 32 + 16 + g * 8);
    }
  }

#pragma unroll
  for (int nt = 0; nt < 4; nt++) {
    int n0 = nblk * 256 + wave * 64 + nt * 16;
    const unsigned short* brow = WAb + (size_t)(n0 + nn) * KP + g * 8;
    v8f acc0 = {}, acc1 = {};
#pragma unroll
    for (int kt = 0; kt < 3; kt++) {
      F16x16 bfr;
      bfr.u.lo = *(const uint4*)(brow + kt * 32);
      bfr.u.hi = *(const uint4*)(brow + kt * 32 + 16);
      acc0 = __builtin_amdgcn_wmma_f32_16x16x32_bf16(
          false, afr[0][kt].v, false, bfr.v, (short)0, acc0, false, false);
      acc1 = __builtin_amdgcn_wmma_f32_16x16x32_bf16(
          false, afr[1][kt].v, false, bfr.v, (short)0, acc1, false, false);
    }
    // C 16x16 f32: lane (16g+n), VGPR r -> element (M = 8g+r, N = n)
    float* out0 = As + (size_t)(m0 + g * 8) * N + n0 + nn;
    float* out1 = As + (size_t)(m0 + 16 + g * 8) * N + n0 + nn;
#pragma unroll
    for (int r = 0; r < 8; r++) out0[(size_t)r * N] = acc0[r];
#pragma unroll
    for (int r = 0; r < 8; r++) out1[(size_t)r * N] = acc1[r];
  }
}

// Persistent per-batch scan over one chunk. 512 threads: output i = tid>>2,
// j-quarter q = tid&3 (32 MACs/lane). Quad partials reduced with __shfl_xor
// (adjacent lanes of one wave32), y ping-pongs between two LDS buffers so
// only ONE barrier per step is needed.
__global__ void __launch_bounds__(512)
k_scan(const float* __restrict__ As, const float* __restrict__ Bs,
       float* __restrict__ ystate, float* __restrict__ out,
       int chunk, int nsteps) {
  __shared__ float ybuf[2][H];
  int b   = blockIdx.x;
  int tid = threadIdx.x;
  int i = tid >> 2, q = tid & 3;

  if (tid < H) ybuf[0][tid] = ystate[b * H + tid];
  __syncthreads();

  for (int t2 = 0; t2 < nsteps; t2++) {
    int t  = chunk * TS + t2;
    int pr = t2 & 1;               // read buffer
    const float* yr   = ybuf[pr];
    const float* arow = As + (size_t)(b * TS + t2) * N + i * H + q * 32;
    if (t2 + 1 < nsteps) {
      // global_prefetch_b8: pull step t+1's 64KB matrix toward the WGP
      __builtin_prefetch(As + (size_t)(b * TS + t2 + 1) * N + tid * 32, 0, 0);
    }
    float partial = 0.f;
#pragma unroll
    for (int k = 0; k < 8; k++) {
      float4 av = *(const float4*)(arow + k * 4);
      float4 yv = *(const float4*)(yr + q * 32 + k * 4);
      partial += av.x * yv.x + av.y * yv.y + av.z * yv.z + av.w * yv.w;
    }
    // in-wave quad reduction: partials for output i sit in lanes i*4 + 0..3
    partial += __shfl_xor(partial, 1, 32);
    partial += __shfl_xor(partial, 2, 32);
    if (q == 0) {
      float st = partial + Bs[((size_t)b * SM1 + t) * H + i];
      float yn = ybuf[pr][i] + CLIP_V * tanhf(st / CLIP_V);
      ybuf[1 - pr][i] = yn;        // write side of ping-pong: no read race
      out[((size_t)b * S + (t + 1)) * H + i] = yn;
    }
    __syncthreads();
  }
  if (tid < H) ystate[b * H + tid] = ybuf[nsteps & 1][tid];
}

extern "C" void kernel_launch(void* const* d_in, const int* in_sizes, int n_in,
                              void* d_out, int out_size, void* d_ws, size_t ws_size,
                              hipStream_t stream) {
  const float* X     = (const float*)d_in[0];   // (8,512,64)
  const float* initW = (const float*)d_in[1];   // (128,64)
  const float* initb = (const float*)d_in[2];   // (128,)
  const float* WA    = (const float*)d_in[3];   // (16384,65)
  const float* WB    = (const float*)d_in[4];   // (128,65)
  float* out = (float*)d_out;                   // (8,512,128)

  // workspace layout (bytes):
  //   [0,      4096)        ystate   (8*128 f32)
  //   [4096,   2097152)     Bs       (8*511*128 f32)
  //   [2097152,2883584)     inpb     (8*512*96 bf16)
  //   [2883584,6029312)     WAb      (16384*96 bf16)
  //   [6029312,39583744)    As chunk (8*64*16384 f32, 33.5MB -> L2-resident)
  char* ws = (char*)d_ws;
  float*          ystate = (float*)(ws);
  float*          Bs     = (float*)(ws + 4096);
  unsigned short* inpb   = (unsigned short*)(ws + 2097152);
  unsigned short* WAb    = (unsigned short*)(ws + 2883584);
  float*          As     = (float*)(ws + 6029312);

  k_init_y0<<<Bx, H, 0, stream>>>(X, initW, initb, ystate, out);
  k_bs<<<(Bx * SM1 * H + 255) / 256, 256, 0, stream>>>(X, WB, Bs);
  k_conv_inp<<<(Bx * S * KP) / 256, 256, 0, stream>>>(X, inpb);
  k_conv_W<<<(N * KP) / 256, 256, 0, stream>>>(WA, WAb);

  for (int c = 0; c < NCH; c++) {
    k_gemm_As<<<1024, 128, 0, stream>>>(inpb, WAb, As, c);
    int nsteps = (c == NCH - 1) ? (SM1 - c * TS) : TS;
    k_scan<<<Bx, 512, 0, stream>>>(As, Bs, ystate, out, c, nsteps);
  }
}
